// MultiHeadAttention_19963007992012
// MI455X (gfx1250) — matrix operand order repaired and checked by reading, not compiled, not run on hardware
//
#include <hip/hip_runtime.h>
#include <stdint.h>

// MultiHeadAttention forward for MI455X (gfx1250), wave32 + WMMA bf16.
// B=2, S=2048, D_MODEL=1024, HEADS=16, D_K=64.

#define D_MODEL 1024
#define HEADS   16
#define DK      64
#define BATCH   2
#define SEQ     2048
#define NROWS   (BATCH * SEQ)   // 4096
#define LDSS    72              // LDS row stride in bf16 elems (64 + 8 pad, 144B rows keep 16B align)

typedef __attribute__((ext_vector_type(16))) __bf16 v16bf;
typedef __attribute__((ext_vector_type(8)))  float  v8f;

// ---------- scalar helpers ----------
__device__ __forceinline__ unsigned short f32_to_bf16(float f) {
  unsigned int u = __float_as_uint(f);
  u += 0x7FFFu + ((u >> 16) & 1u);           // round-to-nearest-even
  return (unsigned short)(u >> 16);
}
__device__ __forceinline__ unsigned int pack_bf16x2(float lo, float hi) {
  return (unsigned int)f32_to_bf16(lo) | ((unsigned int)f32_to_bf16(hi) << 16);
}
__device__ __forceinline__ v8f vzero8() {
  v8f z;
#pragma unroll
  for (int i = 0; i < 8; ++i) z[i] = 0.0f;
  return z;
}
__device__ __forceinline__ float lane_xor_f(float v, int mask, int lane) {
  // cross-lane value from lane^mask (stays inside 16-lane halves for mask<16)
  return __int_as_float(
      __builtin_amdgcn_ds_bpermute(((lane ^ mask) << 2), __float_as_int(v)));
}

// ---------- WMMA fragment loaders (contiguous k = 16*half + s, same for A and B) ----------
// A 16x32 bf16: lane L(0-15)=row M=L, half-lane k base 0/16; VGPR j: k=2j (contiguous, matches B).
__device__ __forceinline__ v16bf load_frag_A(const unsigned short* lds, int rowBase,
                                             int kBase, int lane) {
  const int hlf = lane >> 4;
  const unsigned short* p = lds + (rowBase + (lane & 15)) * LDSS + kBase + hlf * 16;
  union { v16bf v; unsigned int u[8]; } f;
#pragma unroll
  for (int j = 0; j < 4; ++j) f.u[j] = *(const unsigned int*)(p + 2 * j);
#pragma unroll
  for (int j = 0; j < 4; ++j) f.u[4 + j] = *(const unsigned int*)(p + 8 + 2 * j);
  return f.v;
}
// B 32x16 bf16: lane L(0-15)=col N holds K=0..15 (2/VGPR); lanes 16-31 K=16..31.
// LDS tile must hold B as [n][k] with k contiguous.
__device__ __forceinline__ v16bf load_frag_B(const unsigned short* lds, int colBase,
                                             int kBase, int lane) {
  const int hlf = lane >> 4;
  const unsigned short* p = lds + (colBase + (lane & 15)) * LDSS + kBase + hlf * 16;
  union { v16bf v; unsigned int u[8]; } f;
#pragma unroll
  for (int j = 0; j < 8; ++j) f.u[j] = *(const unsigned int*)(p + 2 * j);
  return f.v;
}
__device__ __forceinline__ v8f wmma_bf16(v16bf a, v16bf b, v8f c) {
  return __builtin_amdgcn_wmma_f32_16x16x32_bf16(false, a, false, b, (short)0, c,
                                                 false, false);
}

// ---------- cooperative tile loaders (128 threads, 64x64 tile) ----------
// fp32 global (row stride D_MODEL) -> bf16 LDS [row][col], converting on the fly.
__device__ __forceinline__ void load_tile_f32_bf16(const float* __restrict__ g,
                                                   unsigned short* lds, int tid) {
  const int row = tid >> 1;
  const int c0  = (tid & 1) * 32;
  const float4* src = (const float4*)(g + (size_t)row * D_MODEL + c0);
#pragma unroll
  for (int i = 0; i < 4; ++i) {
    float4 f0 = src[2 * i];
    float4 f1 = src[2 * i + 1];
    uint4 o;
    o.x = pack_bf16x2(f0.x, f0.y);
    o.y = pack_bf16x2(f0.z, f0.w);
    o.z = pack_bf16x2(f1.x, f1.y);
    o.w = pack_bf16x2(f1.z, f1.w);
    *(uint4*)(lds + row * LDSS + c0 + 8 * i) = o;
  }
}

// ---------- GEMM: y[n,m] = sum_d X[n,d]*W[m,d] + bias[m], scaled ----------
// Block: 128 threads (4 waves). Block tile n=128, m=64. Wave tile 32x64 (8 accs).
// mode 0: write bf16 into head layout [B][H][S][DK]; mode 1: write fp32 [n][m].
__global__ __launch_bounds__(128) void gemm_nt_wmma(
    const float* __restrict__ X, const float* __restrict__ W,
    const float* __restrict__ bias, unsigned short* __restrict__ outHead,
    float* __restrict__ outF32, int mode, float outScale) {
  __shared__ unsigned short xs[128 * LDSS];
  __shared__ unsigned short wsm[64 * LDSS];
  const int tid = threadIdx.x;
  const int lane = tid & 31;
  const int wave = tid >> 5;
  const int hlf = lane >> 4;
  const int ln = lane & 15;
  const int m0 = blockIdx.x * 64;
  const int n0 = blockIdx.y * 128;

  v8f acc[2][4];
#pragma unroll
  for (int i = 0; i < 2; ++i)
#pragma unroll
    for (int j = 0; j < 4; ++j) acc[i][j] = vzero8();

  for (int kc = 0; kc < D_MODEL; kc += 64) {
    __syncthreads();
    load_tile_f32_bf16(X + (size_t)n0 * D_MODEL + kc, xs, tid);                  // rows 0..63
    load_tile_f32_bf16(X + (size_t)(n0 + 64) * D_MODEL + kc, xs + 64 * LDSS, tid); // rows 64..127
    load_tile_f32_bf16(W + (size_t)m0 * D_MODEL + kc, wsm, tid);
    __syncthreads();
#pragma unroll
    for (int kk = 0; kk < 2; ++kk) {
      // preload all fragments, then 8 back-to-back WMMAs (one dscnt wait total)
      v16bf a0 = load_frag_A(xs, wave * 32, kk * 32, lane);
      v16bf a1 = load_frag_A(xs, wave * 32 + 16, kk * 32, lane);
      v16bf b0 = load_frag_B(wsm, 0, kk * 32, lane);
      v16bf b1 = load_frag_B(wsm, 16, kk * 32, lane);
      v16bf b2 = load_frag_B(wsm, 32, kk * 32, lane);
      v16bf b3 = load_frag_B(wsm, 48, kk * 32, lane);
      acc[0][0] = wmma_bf16(a0, b0, acc[0][0]);
      acc[1][0] = wmma_bf16(a1, b0, acc[1][0]);
      acc[0][1] = wmma_bf16(a0, b1, acc[0][1]);
      acc[1][1] = wmma_bf16(a1, b1, acc[1][1]);
      acc[0][2] = wmma_bf16(a0, b2, acc[0][2]);
      acc[1][2] = wmma_bf16(a1, b2, acc[1][2]);
      acc[0][3] = wmma_bf16(a0, b3, acc[0][3]);
      acc[1][3] = wmma_bf16(a1, b3, acc[1][3]);
    }
  }

#pragma unroll
  for (int mb = 0; mb < 4; ++mb) {
    const int m = m0 + mb * 16 + ln;
    const float bb = bias[m];
    const int h = m >> 6;
    const int dk = m & 63;
#pragma unroll
    for (int rh = 0; rh < 2; ++rh) {
#pragma unroll
      for (int r = 0; r < 8; ++r) {
        const int n = n0 + wave * 32 + rh * 16 + r + 8 * hlf;  // C layout: row = r + 8*half
        const float val = (acc[rh][mb][r] + bb) * outScale;
        if (mode == 0) {
          const int bt = n >> 11;                 // n / SEQ
          const int s = n & (SEQ - 1);
          const size_t idx = ((((size_t)bt * HEADS + h) * SEQ + s) << 6) + dk;
          outHead[idx] = f32_to_bf16(val);
        } else {
          outF32[(size_t)n * D_MODEL + m] = val;
        }
      }
    }
  }
}

// ---------- Flash attention per (b,h), 64-query tile per block ----------
// Q is pre-scaled by 1/sqrt(DK) in its projection. K/V tiles for step t+1 are
// prefetched into registers while step t computes (hides global latency).
__global__ __launch_bounds__(128) void attn_fwd(
    const unsigned short* __restrict__ Qh, const unsigned short* __restrict__ Kh,
    const unsigned short* __restrict__ Vh, float* __restrict__ att /* [4096][1024] */) {
  __shared__ unsigned short qs[64 * LDSS];
  __shared__ unsigned short ks[64 * LDSS];
  __shared__ unsigned short vts[64 * LDSS];           // V transposed: [d][key]
  __shared__ unsigned short ps[4 * 16 * LDSS];        // per-wave P staging (C->A relayout)

  const int tid = threadIdx.x;
  const int lane = tid & 31;
  const int wave = tid >> 5;
  const int hlf = lane >> 4;
  const int ln = lane & 15;
  const int bh = blockIdx.x;          // b*HEADS + h
  const int bt = bh >> 4;
  const int h = bh & 15;
  const int q0 = blockIdx.y * 64;

  const size_t headBase = (size_t)bh * SEQ * DK;
  const unsigned short* Qp = Qh + headBase + (size_t)q0 * DK;
  const unsigned short* Kp = Kh + headBase;
  const unsigned short* Vp = Vh + headBase;

  const int row = tid >> 1;
  const int c0 = (tid & 1) * 32;

  // Q tile resident in LDS for the whole kernel
  {
    const uint4* s = (const uint4*)(Qp + (size_t)row * DK + c0);
    uint4* d = (uint4*)(qs + row * LDSS + c0);
    d[0] = s[0]; d[1] = s[1]; d[2] = s[2]; d[3] = s[3];
  }

  // prologue: prefetch K/V tile 0 into registers
  uint4 kreg[4], vreg[4];
  {
    const uint4* sk = (const uint4*)(Kp + (size_t)row * DK + c0);
    const uint4* sv = (const uint4*)(Vp + (size_t)row * DK + c0);
#pragma unroll
    for (int i = 0; i < 4; ++i) { kreg[i] = sk[i]; vreg[i] = sv[i]; }
  }

  v8f oacc[4];
#pragma unroll
  for (int i = 0; i < 4; ++i) oacc[i] = vzero8();
  float mi[8], li[8];
#pragma unroll
  for (int r = 0; r < 8; ++r) { mi[r] = -1.0e30f; li[r] = 0.0f; }

  for (int t = 0; t < SEQ / 64; ++t) {
    __syncthreads();                       // previous iteration's LDS reads done
    {                                      // commit prefetched K/V tile to LDS
      uint4* dk_ = (uint4*)(ks + row * LDSS + c0);
      dk_[0] = kreg[0]; dk_[1] = kreg[1]; dk_[2] = kreg[2]; dk_[3] = kreg[3];
      union { uint4 q4[4]; unsigned short s16[32]; } u;
      u.q4[0] = vreg[0]; u.q4[1] = vreg[1]; u.q4[2] = vreg[2]; u.q4[3] = vreg[3];
#pragma unroll
      for (int i = 0; i < 32; ++i) vts[(c0 + i) * LDSS + row] = u.s16[i];
    }
    __syncthreads();
    if (t + 1 < SEQ / 64) {                // prefetch next tile; overlaps compute below
      const uint4* sk = (const uint4*)(Kp + (size_t)(t + 1) * 64 * DK + (size_t)row * DK + c0);
      const uint4* sv = (const uint4*)(Vp + (size_t)(t + 1) * 64 * DK + (size_t)row * DK + c0);
#pragma unroll
      for (int i = 0; i < 4; ++i) { kreg[i] = sk[i]; vreg[i] = sv[i]; }
    }

    // scores S = Qs * K^T  (Q pre-scaled; wave = 16 query rows x 64 key cols)
    v8f sc[4];
#pragma unroll
    for (int i = 0; i < 4; ++i) sc[i] = vzero8();
#pragma unroll
    for (int kk = 0; kk < 2; ++kk) {
      v16bf afr = load_frag_A(qs, wave * 16, kk * 32, lane);
      v16bf b0 = load_frag_B(ks, 0, kk * 32, lane);
      v16bf b1 = load_frag_B(ks, 16, kk * 32, lane);
      v16bf b2 = load_frag_B(ks, 32, kk * 32, lane);
      v16bf b3 = load_frag_B(ks, 48, kk * 32, lane);
      sc[0] = wmma_bf16(afr, b0, sc[0]);
      sc[1] = wmma_bf16(afr, b1, sc[1]);
      sc[2] = wmma_bf16(afr, b2, sc[2]);
      sc[3] = wmma_bf16(afr, b3, sc[3]);
    }

    // online softmax: rowmax over 64 key cols (4 accs x 16-lane half)
    float alpha[8];
#pragma unroll
    for (int r = 0; r < 8; ++r) {
      float v = fmaxf(fmaxf(sc[0][r], sc[1][r]), fmaxf(sc[2][r], sc[3][r]));
      v = fmaxf(v, lane_xor_f(v, 1, lane));
      v = fmaxf(v, lane_xor_f(v, 2, lane));
      v = fmaxf(v, lane_xor_f(v, 4, lane));
      v = fmaxf(v, lane_xor_f(v, 8, lane));
      const float mnew = fmaxf(mi[r], v);
      alpha[r] = __expf(mi[r] - mnew);
      mi[r] = mnew;
    }

    // P = exp(S - m): stage into LDS (C layout -> row-major) and row-sum
    float rsum[8];
#pragma unroll
    for (int r = 0; r < 8; ++r) rsum[r] = 0.0f;
    unsigned short* pw = ps + wave * 16 * LDSS;
#pragma unroll
    for (int a = 0; a < 4; ++a) {
#pragma unroll
      for (int r = 0; r < 8; ++r) {
        const float p = __expf(sc[a][r] - mi[r]);
        rsum[r] += p;
        pw[(r + 8 * hlf) * LDSS + a * 16 + ln] = f32_to_bf16(p);
      }
    }
#pragma unroll
    for (int r = 0; r < 8; ++r) {
      float v = rsum[r];
      v += lane_xor_f(v, 1, lane);
      v += lane_xor_f(v, 2, lane);
      v += lane_xor_f(v, 4, lane);
      v += lane_xor_f(v, 8, lane);
      li[r] = li[r] * alpha[r] + v;
    }
#pragma unroll
    for (int d = 0; d < 4; ++d)
#pragma unroll
      for (int r = 0; r < 8; ++r) oacc[d][r] *= alpha[r];

    __syncthreads();   // P staging visible before A-fragment readback

    // O += P * V   (k-dim = 64 keys, B from transposed V tile)
#pragma unroll
    for (int kk = 0; kk < 2; ++kk) {
      v16bf afr = load_frag_A(pw, 0, kk * 32, lane);
      v16bf b0 = load_frag_B(vts, 0, kk * 32, lane);
      v16bf b1 = load_frag_B(vts, 16, kk * 32, lane);
      v16bf b2 = load_frag_B(vts, 32, kk * 32, lane);
      v16bf b3 = load_frag_B(vts, 48, kk * 32, lane);
      oacc[0] = wmma_bf16(afr, b0, oacc[0]);
      oacc[1] = wmma_bf16(afr, b1, oacc[1]);
      oacc[2] = wmma_bf16(afr, b2, oacc[2]);
      oacc[3] = wmma_bf16(afr, b3, oacc[3]);
    }
  }

  // epilogue: O /= l, write concat layout [b, s, h*64 + d] fp32
  float rl[8];
#pragma unroll
  for (int r = 0; r < 8; ++r) rl[r] = 1.0f / li[r];
#pragma unroll
  for (int d = 0; d < 4; ++d) {
    const int col = h * 64 + d * 16 + ln;
#pragma unroll
    for (int r = 0; r < 8; ++r) {
      const int orow = q0 + wave * 16 + r + 8 * hlf;
      att[(size_t)(bt * SEQ + orow) * D_MODEL + col] = oacc[d][r] * rl[r];
    }
  }
}

// ---------- host ----------
extern "C" void kernel_launch(void* const* d_in, const int* in_sizes, int n_in,
                              void* d_out, int out_size, void* d_ws, size_t ws_size,
                              hipStream_t stream) {
  (void)in_sizes; (void)n_in; (void)out_size;
  const float* q  = (const float*)d_in[0];
  const float* k  = (const float*)d_in[1];
  const float* v  = (const float*)d_in[2];
  const float* Wq = (const float*)d_in[3];
  const float* bq = (const float*)d_in[4];
  const float* Wk = (const float*)d_in[5];
  const float* bk = (const float*)d_in[6];
  const float* Wv = (const float*)d_in[7];
  const float* bv = (const float*)d_in[8];
  const float* Wo = (const float*)d_in[9];
  const float* bo = (const float*)d_in[10];

  const size_t elems = (size_t)NROWS * D_MODEL;   // 4 Mi
  // workspace: Qh/Kh/Vh bf16 (8MB each) + attention output fp32 (16MB) = 40MB
  if (ws_size < elems * 2 * 3 + elems * 4) return;
  char* w = (char*)d_ws;
  unsigned short* Qh = (unsigned short*)w; w += elems * 2;
  unsigned short* Kh = (unsigned short*)w; w += elems * 2;
  unsigned short* Vh = (unsigned short*)w; w += elems * 2;
  float* att = (float*)w;

  dim3 blk(128);
  dim3 gGemm(D_MODEL / 64, NROWS / 128);          // (16, 32)
  const float qscale = 0.125f;                    // 1/sqrt(DK) folded into Q projection
  gemm_nt_wmma<<<gGemm, blk, 0, stream>>>(q, Wq, bq, Qh, nullptr, 0, qscale);
  gemm_nt_wmma<<<gGemm, blk, 0, stream>>>(k, Wk, bk, Kh, nullptr, 0, 1.0f);
  gemm_nt_wmma<<<gGemm, blk, 0, stream>>>(v, Wv, bv, Vh, nullptr, 0, 1.0f);
  attn_fwd<<<dim3(BATCH * HEADS, SEQ / 64), blk, 0, stream>>>(Qh, Kh, Vh, att);
  gemm_nt_wmma<<<gGemm, blk, 0, stream>>>(att, Wo, bo, nullptr, (float*)d_out, 1, 1.0f);
}
